// SATB_12369505812651
// MI455X (gfx1250) — matrix operand order repaired and checked by reading, not compiled, not run on hardware
//
#include <hip/hip_runtime.h>
#include <hip/hip_bf16.h>
#include <math.h>

// ---------------------------------------------------------------------------
// Shapes (fixed by the reference)
// ---------------------------------------------------------------------------
#define Bn   2
#define Tn   64
#define Cn   32
#define Hn   64
#define Wn   64
#define NH   4
#define HID  170
#define Npos 131072          // C*H*W  (GEMM N dimension per batch)
#define Lat  262144          // contiguous token length (2 channels)

typedef __attribute__((ext_vector_type(16))) _Float16 v16h;
typedef __attribute__((ext_vector_type(8)))  float    v8f;
typedef __attribute__((ext_vector_type(4)))  float    v4f;

__device__ __forceinline__ float gelu_exact(float x) {
    return 0.5f * x * (1.0f + erff(x * 0.70710678118654752440f));
}

__device__ __forceinline__ v4f ld4(const float* p) {
    return *reinterpret_cast<const v4f*>(p);
}

// ---------------------------------------------------------------------------
// 1) LayerNorm over channel dim (T) for each (b,c,h,w) position
// ---------------------------------------------------------------------------
__global__ void ln_kernel(const float* __restrict__ x,
                          const float* __restrict__ g,
                          const float* __restrict__ bta,
                          float* __restrict__ out) {
    int pos = blockIdx.x * blockDim.x + threadIdx.x;      // 0 .. 2*Npos-1
    int bb  = pos >> 17;                                  // batch
    int chw = pos & (Npos - 1);
    const float* xp = x   + (size_t)bb * Tn * Npos + chw;
    float*       op = out + (size_t)bb * Tn * Npos + chw;

    float vals[Tn];
    float s = 0.f, s2 = 0.f;
#pragma unroll
    for (int t = 0; t < Tn; ++t) {
        float v = xp[(size_t)t * Npos];
        vals[t] = v; s += v; s2 += v * v;
    }
    float mean = s * (1.f / Tn);
    float var  = s2 * (1.f / Tn) - mean * mean;
    float inv  = rsqrtf(var + 1e-5f);
#pragma unroll
    for (int t = 0; t < Tn; ++t)
        op[(size_t)t * Npos] = (vals[t] - mean) * inv * g[t] + bta[t];
}

// ---------------------------------------------------------------------------
// 2a) WMMA channel-GEMM, K=64 (qkv / attn_out / pin).
//     Out[m,p] = sum_k W[m,k] * B[k,p] (+ residual).
//     No k-guards (K=64 = 2 full chunks). A rows clamped (junk rows never
//     stored), A loaded as 4x b128, B as 16 unconditional dword loads that
//     share one base pointer (offsets (e + 8*(e>>3))*N*4 bytes, max 23*N*4).
// ---------------------------------------------------------------------------
template <bool RES>
__global__ void gemm_k64(const float* __restrict__ W,     // (M,64) row-major
                         const float* __restrict__ Bsrc,  // (64,N) per batch
                         const float* __restrict__ R,     // residual (M,N)
                         float* __restrict__ O,           // (M,N) per batch
                         int M, int n_tiles, int mgroups,
                         long sB, long sO, long sR) {
    const int N = Npos;
    int wave = blockIdx.x * (blockDim.x >> 5) + (threadIdx.x >> 5);
    if (wave >= n_tiles * mgroups) return;    // wave-uniform
    int nt = wave % n_tiles;
    int mg = wave / n_tiles;

    int lane = threadIdx.x & 31;
    int half = lane >> 4;
    int nin  = lane & 15;
    int p    = nt * 16 + nin;

    const float* bp = Bsrc + (size_t)blockIdx.y * sB + p;
    const float* rp = RES ? (R + (size_t)blockIdx.y * sR + p) : nullptr;
    float*       op = O + (size_t)blockIdx.y * sO + p;

    v8f zero = {0.f,0.f,0.f,0.f,0.f,0.f,0.f,0.f};
    v8f acc[4] = {zero, zero, zero, zero};

#pragma unroll
    for (int kc = 0; kc < 2; ++kc) {
        const int kb = kc << 5;
        // ---- B tile 32x16: lane=col, element e holds K = kb+half*8+e (e<8), kb+16+half*8+(e-8) (e>=8) ----
        const float* bcol = bp + (size_t)(kb + (half << 3)) * N;
        float bf[16];
#pragma unroll
        for (int e = 0; e < 16; ++e) bf[e] = bcol[(size_t)(e + ((e >> 3) << 3)) * N];
        v16h bv;
#pragma unroll
        for (int e = 0; e < 16; ++e) bv[e] = (_Float16)bf[e];

        // ---- A tiles: av[0..7] = W[m][kb+half*8 ..], av[8..15] = +16 ----
#pragma unroll
        for (int mt = 0; mt < 4; ++mt) {
            int mbase = (mg * 4 + mt) << 4;
            if (mbase >= M) break;            // wave-uniform
            int mr = mbase + nin;
            if (mr > M - 1) mr = M - 1;       // clamp: junk rows never stored
            const float* wp = W + mr * 64 + kb + (half << 3);
            v4f a0 = ld4(wp);
            v4f a1 = ld4(wp + 4);
            v4f a2 = ld4(wp + 16);
            v4f a3 = ld4(wp + 20);
            v16h av;
#pragma unroll
            for (int i = 0; i < 4; ++i) {
                av[i]      = (_Float16)a0[i];
                av[i + 4]  = (_Float16)a1[i];
                av[i + 8]  = (_Float16)a2[i];
                av[i + 12] = (_Float16)a3[i];
            }
            acc[mt] = __builtin_amdgcn_wmma_f32_16x16x32_f16(
                false, av, false, bv, (short)0, acc[mt], false, false);
        }
    }

    // ---- store: C/D VGPR v -> M = mbase + v + half*8, N = nin ----
#pragma unroll
    for (int mt = 0; mt < 4; ++mt) {
        int mbase = (mg * 4 + mt) << 4;
        if (mbase >= M) break;
        if (mbase + 16 <= M) {                // full tile: unconditional stores
#pragma unroll
            for (int v = 0; v < 8; ++v) {
                int m = mbase + v + (half << 3);
                float val = acc[mt][v];
                if (RES) val += rp[(size_t)m * N];
                op[(size_t)m * N] = val;
            }
        } else {                              // partial last tile (pin M=340)
#pragma unroll
            for (int v = 0; v < 8; ++v) {
                int m = mbase + v + (half << 3);
                if (m < M) {
                    float val = acc[mt][v];
                    if (RES) val += rp[(size_t)m * N];
                    op[(size_t)m * N] = val;
                }
            }
        }
    }
}

// ---------------------------------------------------------------------------
// 2b) Gated WMMA GEMM for pout: M=64, K=HID=170.
//     B[k,p] = gelu(Y[k,p]) * Y[k+170,p]; 5 full chunks (guards fold away),
//     tail chunk uses clamped loads + A-side zero selects (branch-free).
// ---------------------------------------------------------------------------
__global__ void gemm_gated(const float* __restrict__ W,    // (64,170)
                           const float* __restrict__ Y,    // (340,N) per batch
                           const float* __restrict__ R,    // (64,N)
                           float* __restrict__ O,          // (64,N)
                           long sB, long sO, long sR) {
    const int N = Npos, K = HID;
    int wave = blockIdx.x * (blockDim.x >> 5) + (threadIdx.x >> 5);
    if (wave >= 8192) return;
    int nt = wave;

    int lane = threadIdx.x & 31;
    int half = lane >> 4;
    int nin  = lane & 15;
    int p    = nt * 16 + nin;

    const float* yp = Y + (size_t)blockIdx.y * sB + p;
    const float* rp = R + (size_t)blockIdx.y * sR + p;
    float*       op = O + (size_t)blockIdx.y * sO + p;

    v8f zero = {0.f,0.f,0.f,0.f,0.f,0.f,0.f,0.f};
    v8f acc[4] = {zero, zero, zero, zero};

#pragma unroll
    for (int kc = 0; kc < 6; ++kc) {
        const int  kb   = kc << 5;
        const bool full = (kb + 31) < K;      // compile-time after unroll

        v16h bv;
#pragma unroll
        for (int e = 0; e < 16; ++e) {
            int k  = kb + (half << 4) + e;
            int kc2 = full ? k : (k < K ? k : K - 1);   // clamped, in-bounds
            float y1 = yp[(size_t)kc2 * N];
            float y2 = yp[(size_t)(kc2 + HID) * N];
            bv[e] = (_Float16)(gelu_exact(y1) * y2);
        }
#pragma unroll
        for (int mt = 0; mt < 4; ++mt) {
            int m = (mt << 4) + nin;          // M=64: always in range
            v16h av;
#pragma unroll
            for (int e = 0; e < 16; ++e) {
                int v = e >> 1;
                int k = kb + (half << 4) + e;
                bool kok = full || (k < K);
                int  kcl = full ? k : (k < K ? k : K - 1);
                float wv = W[m * K + kcl];
                av[e] = kok ? (_Float16)wv : (_Float16)0.f;   // zero A kills tail
            }
            acc[mt] = __builtin_amdgcn_wmma_f32_16x16x32_f16(
                false, av, false, bv, (short)0, acc[mt], false, false);
        }
    }
#pragma unroll
    for (int mt = 0; mt < 4; ++mt) {
#pragma unroll
        for (int v = 0; v < 8; ++v) {
            int m = (mt << 4) + v + (half << 3);
            op[(size_t)m * N] = acc[mt][v] + rp[(size_t)m * N];
        }
    }
}

// ---------------------------------------------------------------------------
// 3) Depthwise conv, kernel (1,3,3), pad (0,1,1)  -- for qkv
// ---------------------------------------------------------------------------
__global__ void dwconv133(const float* __restrict__ in,
                          const float* __restrict__ w,
                          float* __restrict__ out, int CH) {
    long total = (long)CH * Npos;
    long idx = (long)blockIdx.x * blockDim.x + threadIdx.x;
    if (idx >= total) return;
    int wc = (int)(idx & 63);
    int hr = (int)((idx >> 6) & 63);
    int ch = (int)(idx >> 17);
    size_t boff = (size_t)blockIdx.y * (size_t)total;
    const float* ip = in + boff + (idx - wc - (long)hr * 64);
    const float* wp = w + ch * 9;
    float acc = 0.f;
#pragma unroll
    for (int dh = -1; dh <= 1; ++dh) {
        int hh = hr + dh; if (hh < 0 || hh > 63) continue;
#pragma unroll
        for (int dw = -1; dw <= 1; ++dw) {
            int ww = wc + dw; if (ww < 0 || ww > 63) continue;
            acc += wp[(dh + 1) * 3 + (dw + 1)] * ip[hh * 64 + ww];
        }
    }
    out[boff + idx] = acc;
}

// ---------------------------------------------------------------------------
// 4) Depthwise conv, kernel (3,3,3), pad (1,1,1)  -- for GDFN
// ---------------------------------------------------------------------------
__global__ void dwconv333(const float* __restrict__ in,
                          const float* __restrict__ w,
                          float* __restrict__ out, int CH) {
    long total = (long)CH * Npos;
    long idx = (long)blockIdx.x * blockDim.x + threadIdx.x;
    if (idx >= total) return;
    int wc = (int)(idx & 63);
    int hr = (int)((idx >> 6) & 63);
    int cd = (int)((idx >> 12) & 31);
    int ch = (int)(idx >> 17);
    size_t boff = (size_t)blockIdx.y * (size_t)total;
    const float* ip = in + boff + ((size_t)ch << 17);
    const float* wp = w + ch * 27;
    float acc = 0.f;
#pragma unroll
    for (int dc = -1; dc <= 1; ++dc) {
        int cc = cd + dc; if (cc < 0 || cc > 31) continue;
#pragma unroll
        for (int dh = -1; dh <= 1; ++dh) {
            int hh = hr + dh; if (hh < 0 || hh > 63) continue;
#pragma unroll
            for (int dw = -1; dw <= 1; ++dw) {
                int ww = wc + dw; if (ww < 0 || ww > 63) continue;
                acc += wp[(dc + 1) * 9 + (dh + 1) * 3 + (dw + 1)]
                     * ip[(size_t)cc * 4096 + hh * 64 + ww];
            }
        }
    }
    out[boff + idx] = acc;
}

// ---------------------------------------------------------------------------
// 5) Attention partial sums: per (b,nh), 64 q.k dots + 8 |q|^2 + 8 |k|^2.
//    Token (nh,xi) is the contiguous 262144-float block at channel nh*16+xi*2.
// ---------------------------------------------------------------------------
__global__ void zero_k(float* p, int n) {
    int i = blockIdx.x * blockDim.x + threadIdx.x;
    if (i < n) p[i] = 0.f;
}

__global__ void attn_part(const float* __restrict__ qkv, float* __restrict__ accum) {
    int bn = blockIdx.y;                  // b*4 + nh
    int b  = bn >> 2, nh = bn & 3;
    const float* base = qkv + (size_t)b * 192 * Npos;
    const float* qb = base + (size_t)(nh * 16) * Npos;
    const float* kb = base + (size_t)(64 + nh * 16) * Npos;

    float acc[80];
#pragma unroll
    for (int v = 0; v < 80; ++v) acc[v] = 0.f;

    int lo = blockIdx.x * 8192;
    for (int li = lo + threadIdx.x; li < lo + 8192; li += 256) {
        float qv[8], kv[8];
#pragma unroll
        for (int i = 0; i < 8; ++i) {
            qv[i] = qb[(size_t)i * Lat + li];
            kv[i] = kb[(size_t)i * Lat + li];
        }
#pragma unroll
        for (int i = 0; i < 8; ++i) {
#pragma unroll
            for (int j = 0; j < 8; ++j) acc[i * 8 + j] += qv[i] * kv[j];
            acc[64 + i] += qv[i] * qv[i];
            acc[72 + i] += kv[i] * kv[i];
        }
    }
    __shared__ float red[8][80];
    int lane = threadIdx.x & 31, wv = threadIdx.x >> 5;
#pragma unroll
    for (int v = 0; v < 80; ++v) {
        float s = acc[v];
#pragma unroll
        for (int o = 16; o > 0; o >>= 1) s += __shfl_xor(s, o);
        if (lane == 0) red[wv][v] = s;
    }
    __syncthreads();
    if (threadIdx.x < 80) {
        float s = 0.f;
#pragma unroll
        for (int w = 0; w < 8; ++w) s += red[w][threadIdx.x];
        atomicAdd(&accum[bn * 80 + threadIdx.x], s);
    }
}

// ---------------------------------------------------------------------------
// 6) Finalize: normalize by L2 norms, *temperature, softmax over rows
// ---------------------------------------------------------------------------
__global__ void attn_fin(const float* __restrict__ accum,
                         const float* __restrict__ temp,
                         float* __restrict__ att) {
    int bn = blockIdx.x, nh = bn & 3;
    const float* a = accum + bn * 80;
    __shared__ float sc[8][8], qn[8], kn[8];
    int t = threadIdx.x;
    if (t < 8) {
        qn[t] = fmaxf(sqrtf(a[64 + t]), 1e-12f);
        kn[t] = fmaxf(sqrtf(a[72 + t]), 1e-12f);
    }
    __syncthreads();
    if (t < 64) {
        int i = t >> 3, j = t & 7;
        sc[i][j] = a[i * 8 + j] / (qn[i] * kn[j]) * temp[nh];
    }
    __syncthreads();
    if (t < 8) {
        float mx = -1e30f;
        for (int j = 0; j < 8; ++j) mx = fmaxf(mx, sc[t][j]);
        float e[8], sm = 0.f;
        for (int j = 0; j < 8; ++j) { e[j] = expf(sc[t][j] - mx); sm += e[j]; }
        for (int j = 0; j < 8; ++j) att[bn * 64 + t * 8 + j] = e[j] / sm;
    }
}

// ---------------------------------------------------------------------------
// 7) o = att @ v  (8-wide recombination, v loaded once per element)
// ---------------------------------------------------------------------------
__global__ void attn_apply(const float* __restrict__ qkv,
                           const float* __restrict__ att,
                           float* __restrict__ o) {
    int bn = blockIdx.y, b = bn >> 2, nh = bn & 3;
    __shared__ float a[64];
    if (threadIdx.x < 64) a[threadIdx.x] = att[bn * 64 + threadIdx.x];
    __syncthreads();
    int li = blockIdx.x * 256 + threadIdx.x;
    const float* vb = qkv + (size_t)b * 192 * Npos + (size_t)(128 + nh * 16) * Npos;
    float* ob = o + (size_t)b * 64 * Npos + (size_t)(nh * 16) * Npos;
    float vv[8];
#pragma unroll
    for (int j = 0; j < 8; ++j) vv[j] = vb[(size_t)j * Lat + li];
#pragma unroll
    for (int i = 0; i < 8; ++i) {
        float s = 0.f;
#pragma unroll
        for (int j = 0; j < 8; ++j) s += a[i * 8 + j] * vv[j];
        ob[(size_t)i * Lat + li] = s;
    }
}

// ---------------------------------------------------------------------------
// 8) SE: pool, sigmoid(W*mean+b), scale
// ---------------------------------------------------------------------------
__global__ void pool_k(const float* __restrict__ x, float* __restrict__ mean) {
    __shared__ float red[256];
    int bt = blockIdx.x;
    const float* p = x + (size_t)bt * Npos;
    float s = 0.f;
    for (int i = threadIdx.x; i < Npos; i += 256) s += p[i];
    red[threadIdx.x] = s; __syncthreads();
    for (int o = 128; o > 0; o >>= 1) {
        if (threadIdx.x < o) red[threadIdx.x] += red[threadIdx.x + o];
        __syncthreads();
    }
    if (threadIdx.x == 0) mean[bt] = red[0] * (1.f / Npos);
}

__global__ void se_k(const float* __restrict__ mean,
                     const float* __restrict__ sw,
                     const float* __restrict__ sb,
                     float* __restrict__ sig) {
    int o = threadIdx.x;                 // 0..127
    if (o < 128) {
        int b = o >> 6, oc = o & 63;
        float s = sb[oc];
        for (int t = 0; t < Tn; ++t) s += sw[oc * Tn + t] * mean[b * Tn + t];
        sig[o] = 1.f / (1.f + expf(-s));
    }
}

__global__ void scale_k(const float* __restrict__ x,
                        const float* __restrict__ sig,
                        float* __restrict__ out) {
    size_t idx = (size_t)blockIdx.x * 256 + threadIdx.x;
    int bt = (int)(idx >> 17);
    out[idx] = sig[bt] * x[idx];
}

// ---------------------------------------------------------------------------
// Launcher
// ---------------------------------------------------------------------------
extern "C" void kernel_launch(void* const* d_in, const int* in_sizes, int n_in,
                              void* d_out, int out_size, void* d_ws, size_t ws_size,
                              hipStream_t stream) {
    (void)in_sizes; (void)n_in; (void)out_size; (void)ws_size;
    const float* x        = (const float*)d_in[0];
    const float* n1g      = (const float*)d_in[1];
    const float* n1b      = (const float*)d_in[2];
    const float* temp     = (const float*)d_in[3];
    const float* qkv_w    = (const float*)d_in[4];
    const float* qkv_dw   = (const float*)d_in[5];
    const float* aout_w   = (const float*)d_in[6];
    const float* n2g      = (const float*)d_in[7];
    const float* n2b      = (const float*)d_in[8];
    const float* pin_w    = (const float*)d_in[9];
    const float* dw_w     = (const float*)d_in[10];
    const float* pout_w   = (const float*)d_in[11];
    const float* se_w     = (const float*)d_in[12];
    const float* se_b     = (const float*)d_in[13];
    float* out = (float*)d_out;

    // workspace layout (floats)
    float* ws    = (float*)d_ws;
    float* t0    = ws;                         // 2*340*Npos = 89,128,960
    float* t1    = t0 + 89128960L;             // 89,128,960
    float* xn    = t1 + 89128960L;             // 16,777,216 (LN out, later x2)
    float* xA    = xn + 16777216L;             // 16,777,216 (post-attn residual)
    float* accum = xA + 16777216L;             // 640
    float* att   = accum + 640;                // 512
    float* meanb = att + 512;                  // 128
    float* sig   = meanb + 128;                // 128

    const long sX   = (long)Tn * Npos;          // 64 * 131072
    const long sQKV = 192L * Npos;
    const long sPIN = 340L * Npos;

    // ---- attention branch ----
    ln_kernel<<<1024, 256, 0, stream>>>(x, n1g, n1b, xn);
    gemm_k64<false><<<dim3(3072, 2), 256, 0, stream>>>(
        qkv_w, xn, nullptr, t0, 192, 8192, 3, sX, sQKV, 0);
    dwconv133<<<dim3(98304, 2), 256, 0, stream>>>(t0, qkv_dw, t1, 192);
    zero_k<<<3, 256, 0, stream>>>(accum, 640);
    attn_part<<<dim3(32, 8), 256, 0, stream>>>(t1, accum);
    attn_fin<<<8, 64, 0, stream>>>(accum, temp, att);
    attn_apply<<<dim3(1024, 8), 256, 0, stream>>>(t1, att, t0);   // o -> t0
    gemm_k64<true><<<dim3(1024, 2), 256, 0, stream>>>(
        aout_w, t0, x, xA, 64, 8192, 1, sX, sX, sX);

    // ---- GDFN branch ----
    ln_kernel<<<1024, 256, 0, stream>>>(xA, n2g, n2b, xn);
    gemm_k64<false><<<dim3(6144, 2), 256, 0, stream>>>(
        pin_w, xn, nullptr, t0, 340, 8192, 6, sX, sPIN, 0);
    dwconv333<<<dim3(174080, 2), 256, 0, stream>>>(t0, dw_w, t1, 340);
    gemm_gated<<<dim3(1024, 2), 256, 0, stream>>>(
        pout_w, t1, xA, xn /*x2*/, sPIN, sX, sX);

    // ---- SE gate ----
    pool_k<<<128, 256, 0, stream>>>(xn, meanb);
    se_k<<<1, 128, 0, stream>>>(meanb, se_w, se_b, sig);
    scale_k<<<65536, 256, 0, stream>>>(xn, sig, out);
}
